// Seq2SeqAttentionDecoder_55473797595833
// MI455X (gfx1250) — matrix operand order repaired and checked by reading, compiled, not run on hardware
//
#include <hip/hip_runtime.h>
#include <hip/hip_bf16.h>
#include <math.h>

// ---------------------------------------------------------------------------
// Problem constants (match reference)
// ---------------------------------------------------------------------------
#define V_SZ   32000
#define E_SZ   256
#define H_SZ   512
#define B_SZ   64
#define T_DEC  64
#define S_ENC  128

typedef __attribute__((ext_vector_type(16))) __bf16 bf16x16_t;
typedef __attribute__((ext_vector_type(8)))  __bf16 bf16x8_t;
typedef __attribute__((ext_vector_type(8)))  float  f32x8_t;

// ---------------------------------------------------------------------------
// fp32 -> bf16 conversion (grid-stride)
// ---------------------------------------------------------------------------
__global__ void cvt_f32_bf16_kernel(const float* __restrict__ src,
                                    __bf16* __restrict__ dst, int n) {
    for (int i = blockIdx.x * blockDim.x + threadIdx.x; i < n;
         i += gridDim.x * blockDim.x)
        dst[i] = (__bf16)src[i];
}

// ---------------------------------------------------------------------------
// Pack GRU weights: dst [2048, Kc] bf16 where Kc = Kih + 512.
// Row groups of 512: [r: Wih_r|Whh_r] [z: Wih_z|Whh_z] [n_i: Wih_n|0] [n_h: 0|Whh_n]
// ---------------------------------------------------------------------------
__global__ void pack_w_kernel(const float* __restrict__ Wih,
                              const float* __restrict__ Whh,
                              __bf16* __restrict__ dst, int Kih, int Kc) {
    int total = 2048 * Kc;
    for (int idx = blockIdx.x * blockDim.x + threadIdx.x; idx < total;
         idx += gridDim.x * blockDim.x) {
        int row = idx / Kc, col = idx % Kc;
        int g = row >> 9;          // gate chunk 0..3
        int j = row & 511;
        float v;
        if (g <= 1) {
            int srow = g * 512 + j;
            v = (col < Kih) ? Wih[(size_t)srow * Kih + col]
                            : Whh[(size_t)srow * 512 + (col - Kih)];
        } else if (g == 2) {
            v = (col < Kih) ? Wih[(size_t)(1024 + j) * Kih + col] : 0.0f;
        } else {
            v = (col >= Kih) ? Whh[(size_t)(1024 + j) * 512 + (col - Kih)] : 0.0f;
        }
        dst[idx] = (__bf16)v;
    }
}

// Pack biases to match: [bih_r+bhh_r, bih_z+bhh_z, bih_n, bhh_n]  (2048 floats)
__global__ void pack_b_kernel(const float* __restrict__ bih,
                              const float* __restrict__ bhh,
                              float* __restrict__ dst) {
    int idx = blockIdx.x * blockDim.x + threadIdx.x;
    if (idx >= 2048) return;
    int g = idx >> 9, j = idx & 511;
    float v = (g == 0) ? bih[j] + bhh[j]
            : (g == 1) ? bih[512 + j] + bhh[512 + j]
            : (g == 2) ? bih[1024 + j]
                       : bhh[1024 + j];
    dst[idx] = v;
}

// Seed bf16 copies of the initial hidden state into the concat activation rows
__global__ void init_hbf_kernel(const float* __restrict__ hs,  // [2,64,512]
                                __bf16* __restrict__ acat0,    // [64,1280]
                                __bf16* __restrict__ acat1) {  // [64,1024]
    int idx = blockIdx.x * blockDim.x + threadIdx.x;
    if (idx >= 2 * B_SZ * H_SZ) return;
    int l = idx >> 15;
    int r = idx & 32767;
    int b = r >> 9, j = r & 511;
    __bf16 v = (__bf16)hs[idx];
    if (l == 0) acat0[b * 1280 + 768 + j] = v;
    else        acat1[b * 1024 + 512 + j] = v;
}

// ---------------------------------------------------------------------------
// bf16 WMMA GEMM with async-LDS B staging:
//   C[M,N](f32) = A[M,K](bf16,row-major,lda) @ W[N,K]^T (ldw)
// Block = 128 threads (4 waves), 64x64 C tile; wave w owns M strip 16*w.
// The 64xK B panel is shared by all 4 waves, so it is staged through LDS in
// K-chunks of 64 via GLOBAL_LOAD_ASYNC_TO_LDS_B128 (ASYNCcnt), double
// buffered: while chunk k is consumed by WMMAs, chunk k+1 is in flight
// (s_wait_asynccnt 0x4 keeps 4 loads outstanding per wave).
// LDS row stride = 72 bf16 (144 B) to spread banks.
// Fragment layouts follow CDNA5 ISA 7.12.2 (wave32).
// Optional bias[N]; optional output permute row m = t*permB + b ->
// C[(b*permT + t)*N + n]  (final [B,T,V] logits layout).
// ---------------------------------------------------------------------------
#define KCH      64
#define LDS_ROW  72                      // bf16 elems per LDS row (144 B)
#define LDS_BUF  (64 * LDS_ROW)          // one chunk buffer (bf16 elems)

__global__ void gemm_bf16_wmma_kernel(const __bf16* __restrict__ A, int lda,
                                      const __bf16* __restrict__ W, int ldw,
                                      float* __restrict__ C,
                                      int M, int N, int K,
                                      const float* __restrict__ bias,
                                      int permB, int permT) {
    __shared__ __align__(16) __bf16 ldsB[2 * LDS_BUF];

    const int tid  = threadIdx.x;
    const int wave = tid >> 5;
    const int lane = tid & 31;
    const int l16  = lane & 15;
    const int lhi  = lane >> 4;                 // 0 or 1
    const int m0   = blockIdx.y * 64 + wave * 16;
    const int n0   = blockIdx.x * 64;

    // ---- per-thread async-copy mapping: 512 x b128 per chunk, 4 per lane ----
    // id = tid + 128*i ; row = id>>3 (0..63), seg = id&7 (16B segment in row)
    const uint32_t lbase = (uint32_t)(uintptr_t)&ldsB[0];
    const uint64_t wbase = (uint64_t)(uintptr_t)W;
    uint32_t goff[4], loff[4];
    #pragma unroll
    for (int i = 0; i < 4; ++i) {
        int id = tid + 128 * i;
        int row = id >> 3, seg = id & 7;
        goff[i] = (uint32_t)(((size_t)(n0 + row) * ldw + seg * 8) * 2);
        loff[i] = lbase + (uint32_t)(row * (LDS_ROW * 2) + seg * 16);
    }

    auto issue_chunk = [&](int kc, int buf) {
        #pragma unroll
        for (int i = 0; i < 4; ++i) {
            uint32_t g = goff[i] + (uint32_t)(kc * 2);
            uint32_t l = loff[i] + (uint32_t)(buf * (LDS_BUF * 2));
            asm volatile("global_load_async_to_lds_b128 %0, %1, %2"
                         :: "v"(l), "v"(g), "s"(wbase) : "memory");
        }
    };

    f32x8_t acc[4];
    #pragma unroll
    for (int i = 0; i < 4; ++i)
        #pragma unroll
        for (int r = 0; r < 8; ++r) acc[i][r] = 0.0f;

    const __bf16* arow = A + (size_t)(m0 + l16) * lda;

    issue_chunk(0, 0);
    int buf = 0;
    for (int kc = 0; kc < K; kc += KCH) {
        const bool more = (kc + KCH) < K;
        if (more) {
            issue_chunk(kc + KCH, buf ^ 1);
            asm volatile("s_wait_asynccnt 0x4" ::: "memory");
        } else {
            asm volatile("s_wait_asynccnt 0x0" ::: "memory");
        }
        __syncthreads();                       // chunk visible to all 4 waves

        const __bf16* bbuf = &ldsB[buf * LDS_BUF];
        #pragma unroll
        for (int kk = 0; kk < 2; ++kk) {       // two K=32 WMMA steps per chunk
            const int k0 = kc + kk * 32;
            // A fragment (registers, from global; speculative prefetch ahead)
            const __bf16* pa = arow + k0 + lhi * 8;
            __builtin_prefetch(pa + KCH, 0, 1);          // global_prefetch_b8
            bf16x8_t alo = *reinterpret_cast<const bf16x8_t*>(pa);
            bf16x8_t ahi = *reinterpret_cast<const bf16x8_t*>(pa + 16);
            bf16x16_t afrag = __builtin_shufflevector(alo, ahi,
                0, 1, 2, 3, 4, 5, 6, 7, 8, 9, 10, 11, 12, 13, 14, 15);
            // 4 B fragments from LDS (ds_load_b128 x2 each) + WMMA
            #pragma unroll
            for (int i = 0; i < 4; ++i) {
                const __bf16* pb = bbuf + (i * 16 + l16) * LDS_ROW
                                        + kk * 32 + lhi * 8;
                bf16x8_t b0 = *reinterpret_cast<const bf16x8_t*>(pb);
                bf16x8_t b1 = *reinterpret_cast<const bf16x8_t*>(pb + 16);
                bf16x16_t bfrag = __builtin_shufflevector(b0, b1,
                    0, 1, 2, 3, 4, 5, 6, 7, 8, 9, 10, 11, 12, 13, 14, 15);
                acc[i] = __builtin_amdgcn_wmma_f32_16x16x32_bf16(
                    false, afrag, false, bfrag, (short)0, acc[i], false, false);
            }
        }
        __syncthreads();                       // all waves done before overwrite
        buf ^= 1;
    }

    // ---- store ----
    #pragma unroll
    for (int i = 0; i < 4; ++i) {
        #pragma unroll
        for (int r = 0; r < 8; ++r) {
            int m = m0 + lhi * 8 + r;
            int n = n0 + i * 16 + l16;
            float v = acc[i][r];
            if (bias) v += bias[n];
            size_t oidx;
            if (permT) {
                int b = m % permB, t = m / permB;
                oidx = ((size_t)b * permT + t) * (size_t)N + n;
            } else {
                oidx = (size_t)m * N + n;
            }
            C[oidx] = v;
        }
    }
}

// ---------------------------------------------------------------------------
// Fused additive-attention step (one workgroup per batch row, 128 threads):
//   q = h1[b] @ Wq^T ; scores = wv . tanh(q + keys[b,s,:]) ; masked softmax ;
//   ctx = attn @ enc[b] ; write [ctx | emb[X[b,t]]] as bf16 into acat0 row.
// ---------------------------------------------------------------------------
__global__ void attn_step_kernel(const float* __restrict__ h1,     // [B,H]
                                 const float* __restrict__ Wq,     // [H,H]
                                 const float* __restrict__ keys,   // [B,S,H]
                                 const float* __restrict__ wv,     // [H]
                                 const float* __restrict__ enc,    // [B,S,H]
                                 const int*   __restrict__ vlen,   // [B]
                                 const int*   __restrict__ X,      // [B,T]
                                 const float* __restrict__ emb,    // [V,E]
                                 __bf16* __restrict__ acat0,       // [B,1280]
                                 int t) {
    __shared__ float sh_h[H_SZ];
    __shared__ float sh_q[H_SZ];
    __shared__ float sh_sc[S_ENC];
    __shared__ float sh_red[S_ENC];

    const int b   = blockIdx.x;
    const int tid = threadIdx.x;           // 0..127

    // stage 0: h1[b] -> LDS
    for (int k = tid; k < H_SZ; k += 128) sh_h[k] = h1[b * H_SZ + k];
    __syncthreads();

    // stage 1: q[j] = sum_k h1[k] * Wq[j,k]
    for (int jj = 0; jj < 4; ++jj) {
        int j = tid * 4 + jj;
        const float* wrow = Wq + (size_t)j * H_SZ;
        float s = 0.0f;
        for (int k = 0; k < H_SZ; ++k) s = fmaf(sh_h[k], wrow[k], s);
        sh_q[j] = s;
    }
    __syncthreads();

    // stage 2: scores (one per thread) + mask
    {
        int s = tid;
        const float* krow = keys + ((size_t)b * S_ENC + s) * H_SZ;
        float sc = 0.0f;
        for (int j = 0; j < H_SZ; ++j)
            sc = fmaf(wv[j], tanhf(sh_q[j] + krow[j]), sc);
        if (s >= vlen[b]) sc = -1e6f;
        sh_sc[s] = sc;
        sh_red[s] = sc;
    }
    __syncthreads();

    // stage 3: softmax over 128
    for (int off = 64; off > 0; off >>= 1) {
        if (tid < off) sh_red[tid] = fmaxf(sh_red[tid], sh_red[tid + off]);
        __syncthreads();
    }
    float mx = sh_red[0];
    __syncthreads();
    float e = __expf(sh_sc[tid] - mx);
    sh_sc[tid]  = e;
    sh_red[tid] = e;
    __syncthreads();
    for (int off = 64; off > 0; off >>= 1) {
        if (tid < off) sh_red[tid] += sh_red[tid + off];
        __syncthreads();
    }
    float inv = 1.0f / sh_red[0];
    __syncthreads();
    sh_sc[tid] *= inv;                     // attn weights
    __syncthreads();

    // stage 4: ctx[h] = sum_s attn[s] * enc[b,s,h]  (coalesced over h)
    __bf16* orow = acat0 + (size_t)b * 1280;
    for (int hh = 0; hh < 4; ++hh) {
        int h = tid + 128 * hh;
        const float* ecol = enc + (size_t)b * S_ENC * H_SZ + h;
        float c = 0.0f;
        for (int s = 0; s < S_ENC; ++s)
            c = fmaf(sh_sc[s], ecol[(size_t)s * H_SZ], c);
        orow[h] = (__bf16)c;
    }

    // stage 5: embedding slice
    int tok = X[b * T_DEC + t];
    const float* erow = emb + (size_t)tok * E_SZ;
    for (int e2 = tid; e2 < E_SZ; e2 += 128)
        orow[H_SZ + e2] = (__bf16)erow[e2];
}

// ---------------------------------------------------------------------------
// GRU elementwise: gates [64,2048] f32 = [r_sum, z_sum, n_i, n_h] chunks.
//   r = sig(g0+bb0); z = sig(g1+bb1); n = tanh(g2+bb2 + r*(g3+bb3));
//   h' = (1-z)*n + z*h_prev.  Writes h' f32 + two bf16 copies (strided rows).
// ---------------------------------------------------------------------------
__global__ void gru_elem_kernel(const float* __restrict__ gates,
                                const float* __restrict__ bb,     // [2048]
                                const float* __restrict__ hprev,  // [64,512]
                                float* __restrict__ hnew,         // [64,512]
                                __bf16* __restrict__ d1, int ld1,
                                __bf16* __restrict__ d2, int ld2) {
    int idx = blockIdx.x * blockDim.x + threadIdx.x;
    if (idx >= B_SZ * H_SZ) return;
    int b = idx >> 9, j = idx & 511;
    const float* g = gates + (size_t)b * 2048;
    float r = 1.0f / (1.0f + __expf(-(g[j]        + bb[j])));
    float z = 1.0f / (1.0f + __expf(-(g[512 + j]  + bb[512 + j])));
    float n = tanhf(g[1024 + j] + bb[1024 + j] + r * (g[1536 + j] + bb[1536 + j]));
    float h = (1.0f - z) * n + z * hprev[idx];
    hnew[idx] = h;
    __bf16 hb = (__bf16)h;
    d1[(size_t)b * ld1 + j] = hb;
    d2[(size_t)b * ld2 + j] = hb;
}

// ---------------------------------------------------------------------------
// Host orchestration
// ---------------------------------------------------------------------------
extern "C" void kernel_launch(void* const* d_in, const int* in_sizes, int n_in,
                              void* d_out, int out_size, void* d_ws, size_t ws_size,
                              hipStream_t stream) {
    const int*   X     = (const int*)  d_in[0];
    const float* enc   = (const float*)d_in[1];
    const float* hs0   = (const float*)d_in[2];
    const int*   vlen  = (const int*)  d_in[3];
    const float* emb   = (const float*)d_in[4];
    const float* Wq    = (const float*)d_in[5];
    const float* Wk    = (const float*)d_in[6];
    const float* wv    = (const float*)d_in[7];
    const float* Wih0  = (const float*)d_in[8];
    const float* Whh0  = (const float*)d_in[9];
    const float* bih0  = (const float*)d_in[10];
    const float* bhh0  = (const float*)d_in[11];
    const float* Wih1  = (const float*)d_in[12];
    const float* Whh1  = (const float*)d_in[13];
    const float* bih1  = (const float*)d_in[14];
    const float* bhh1  = (const float*)d_in[15];
    const float* Wd    = (const float*)d_in[16];
    const float* bd    = (const float*)d_in[17];
    float*       out   = (float*)d_out;

    // workspace carve-up (256B aligned blocks)
    char* wp = (char*)d_ws;
    auto carve = [&](size_t bytes) -> void* {
        void* p = (void*)wp;
        wp += (bytes + 255) & ~(size_t)255;
        return p;
    };
    __bf16* enc_bf  = (__bf16*)carve((size_t)B_SZ * S_ENC * H_SZ * 2);  // 8 MB
    __bf16* Wk_bf   = (__bf16*)carve((size_t)H_SZ * H_SZ * 2);
    float*  keys    = (float*) carve((size_t)B_SZ * S_ENC * H_SZ * 4);  // 16 MB
    __bf16* Wp0     = (__bf16*)carve((size_t)2048 * 1280 * 2);
    __bf16* Wp1     = (__bf16*)carve((size_t)2048 * 1024 * 2);
    float*  bp0     = (float*) carve(2048 * 4);
    float*  bp1     = (float*) carve(2048 * 4);
    __bf16* Wd_bf   = (__bf16*)carve((size_t)V_SZ * H_SZ * 2);          // 33 MB
    float*  h_f32   = (float*) carve((size_t)2 * B_SZ * H_SZ * 4);
    __bf16* acat0   = (__bf16*)carve((size_t)B_SZ * 1280 * 2);
    __bf16* acat1   = (__bf16*)carve((size_t)B_SZ * 1024 * 2);
    float*  gates   = (float*) carve((size_t)B_SZ * 2048 * 4);
    __bf16* outs_bf = (__bf16*)carve((size_t)T_DEC * B_SZ * H_SZ * 2);  // 4 MB

    // ---- one-time conversions / packing ----
    cvt_f32_bf16_kernel<<<2048, 256, 0, stream>>>(enc, enc_bf, B_SZ * S_ENC * H_SZ);
    cvt_f32_bf16_kernel<<<512,  256, 0, stream>>>(Wk,  Wk_bf,  H_SZ * H_SZ);
    cvt_f32_bf16_kernel<<<4096, 256, 0, stream>>>(Wd,  Wd_bf,  V_SZ * H_SZ);
    pack_w_kernel<<<2048, 256, 0, stream>>>(Wih0, Whh0, Wp0, 768, 1280);
    pack_w_kernel<<<2048, 256, 0, stream>>>(Wih1, Whh1, Wp1, 512, 1024);
    pack_b_kernel<<<8, 256, 0, stream>>>(bih0, bhh0, bp0);
    pack_b_kernel<<<8, 256, 0, stream>>>(bih1, bhh1, bp1);
    hipMemcpyAsync(h_f32, hs0, (size_t)2 * B_SZ * H_SZ * sizeof(float),
                   hipMemcpyDeviceToDevice, stream);
    init_hbf_kernel<<<(2 * B_SZ * H_SZ + 255) / 256, 256, 0, stream>>>(hs0, acat0, acat1);

    // ---- keys = enc @ Wk^T : [8192,512] = [8192,512] x [512,512]^T ----
    gemm_bf16_wmma_kernel<<<dim3(H_SZ / 64, (B_SZ * S_ENC) / 64), 128, 0, stream>>>(
        enc_bf, H_SZ, Wk_bf, H_SZ, keys, B_SZ * S_ENC, H_SZ, H_SZ, nullptr, 0, 0);

    float* h0 = h_f32;
    float* h1 = h_f32 + B_SZ * H_SZ;

    // ---- recurrent decode ----
    for (int t = 0; t < T_DEC; ++t) {
        attn_step_kernel<<<B_SZ, 128, 0, stream>>>(
            h1, Wq, keys, wv, enc, vlen, X, emb, acat0, t);

        // layer 0 gates: [64,2048] = acat0[64,1280] @ Wp0^T
        gemm_bf16_wmma_kernel<<<dim3(2048 / 64, 1), 128, 0, stream>>>(
            acat0, 1280, Wp0, 1280, gates, B_SZ, 2048, 1280, nullptr, 0, 0);
        gru_elem_kernel<<<(B_SZ * H_SZ + 255) / 256, 256, 0, stream>>>(
            gates, bp0, h0, h0, acat0 + 768, 1280, acat1, 1024);

        // layer 1 gates: [64,2048] = acat1[64,1024] @ Wp1^T
        gemm_bf16_wmma_kernel<<<dim3(2048 / 64, 1), 128, 0, stream>>>(
            acat1, 1024, Wp1, 1024, gates, B_SZ, 2048, 1024, nullptr, 0, 0);
        gru_elem_kernel<<<(B_SZ * H_SZ + 255) / 256, 256, 0, stream>>>(
            gates, bp1, h1, h1, acat1 + 512, 1024,
            outs_bf + (size_t)t * B_SZ * H_SZ, H_SZ);
    }

    // ---- logits: [4096,32000] = outs_bf @ Wd^T + bd, permuted to [B,T,V] ----
    gemm_bf16_wmma_kernel<<<dim3(V_SZ / 64, (T_DEC * B_SZ) / 64), 128, 0, stream>>>(
        outs_bf, H_SZ, Wd_bf, H_SZ, out, T_DEC * B_SZ, V_SZ, H_SZ, bd, B_SZ, T_DEC);
    (void)in_sizes; (void)n_in; (void)out_size; (void)ws_size;
}
